// classifier_29076928594298
// MI455X (gfx1250) — matrix operand order repaired and checked by reading, not compiled, not run on hardware
//
#include <hip/hip_runtime.h>
#include <hip/hip_bf16.h>
#include <stdint.h>

// ---------------------------------------------------------------------------
// CDNA5 (gfx1250) GATv2 GNN pipeline.  Wave32.
// GEMMs: v_wmma_f32_16x16x32_bf16 with async global->LDS weight staging.
// ---------------------------------------------------------------------------

typedef __attribute__((ext_vector_type(16))) __bf16 v16bf;
typedef __attribute__((ext_vector_type(8)))  float  v8f;

#define HID        128
#define EDIM       64
#define NGRAPH     32
#define NEG_SLOPE  0.2f
#define BN_EPS     1e-5f
#define KCHUNK     64          // K-chunk staged in LDS per iteration

// ----------------------------- device helpers ------------------------------

__device__ __forceinline__ unsigned short f32_to_bf16(float f) {
  unsigned u = __float_as_uint(f);
  unsigned r = 0x7fffu + ((u >> 16) & 1u);          // round-to-nearest-even
  return (unsigned short)((u + r) >> 16);
}
__device__ __forceinline__ float bf16_to_f32(unsigned bits16) {
  return __uint_as_float(bits16 << 16);
}
// monotonic float<->uint mapping so atomicMax(u32) == float max
__device__ __forceinline__ unsigned f2mono(float f) {
  unsigned u = __float_as_uint(f);
  return (u & 0x80000000u) ? ~u : (u | 0x80000000u);
}
__device__ __forceinline__ float mono2f(unsigned u) {
  return (u & 0x80000000u) ? __uint_as_float(u & 0x7fffffffu)
                           : __uint_as_float(~u);
}
__device__ __forceinline__ float lrelu(float x) { return x >= 0.f ? x : NEG_SLOPE * x; }

// ------------------------------ small kernels ------------------------------

__global__ void fill_u32_kernel(unsigned* __restrict__ p, unsigned v, size_t n) {
  size_t i = (size_t)blockIdx.x * blockDim.x + threadIdx.x;
  if (i < n) p[i] = v;
}

__global__ void f32_to_bf16_kernel(const float* __restrict__ s,
                                   unsigned short* __restrict__ d, size_t n) {
  size_t i = (size_t)blockIdx.x * blockDim.x + threadIdx.x;
  if (i < n) d[i] = f32_to_bf16(s[i]);
}

// copy [N,C] f32 block into column window [colOff, colOff+C) of [N,dstC] bf16
__global__ void cat_bf16_kernel(const float* __restrict__ src,
                                unsigned short* __restrict__ dst,
                                int N, int C, int dstC, int colOff) {
  size_t i = (size_t)blockIdx.x * blockDim.x + threadIdx.x;
  size_t total = (size_t)N * C;
  if (i >= total) return;
  int n = (int)(i / C);
  int c = (int)(i % C);
  dst[(size_t)n * dstC + colOff + c] = f32_to_bf16(src[i]);
}

// append self-loop edges:  srcf/dstf[0..E) = edge_index rows, [E..E+N) = loops
__global__ void build_edges_kernel(const int* __restrict__ ei,
                                   int* __restrict__ srcf, int* __restrict__ dstf,
                                   int E, int N) {
  int i = blockIdx.x * blockDim.x + threadIdx.x;
  if (i < E)              { srcf[i] = ei[i];      dstf[i] = ei[E + i]; }
  else if (i < E + N)     { srcf[i] = i - E;      dstf[i] = i - E; }
}

__global__ void cnt_kernel(const int* __restrict__ ei, float* __restrict__ cnt, int E) {
  int e = blockIdx.x * blockDim.x + threadIdx.x;
  if (e < E) atomicAdd(&cnt[ei[E + e]], 1.f);
}

// one wave per edge; 64 edge-attr channels, 2 per lane
__global__ void loop_sum_kernel(const int* __restrict__ ei, const float* __restrict__ ea,
                                float* __restrict__ lsum, int E) {
  int e = blockIdx.x * (blockDim.x >> 5) + (threadIdx.x >> 5);
  if (e >= E) return;
  int lane = threadIdx.x & 31;
  int d = ei[E + e];
  atomicAdd(&lsum[(size_t)d * EDIM + lane],      ea[(size_t)e * EDIM + lane]);
  atomicAdd(&lsum[(size_t)d * EDIM + lane + 32], ea[(size_t)e * EDIM + lane + 32]);
}

// loop_attr = lsum / max(cnt,1), written (bf16) into the self-loop tail of ea_bf
__global__ void loopdiv_kernel(const float* __restrict__ lsum, const float* __restrict__ cnt,
                               unsigned short* __restrict__ eabf, int N, int E) {
  size_t i = (size_t)blockIdx.x * blockDim.x + threadIdx.x;
  if (i >= (size_t)N * EDIM) return;
  int n = (int)(i >> 6);
  float c = cnt[n]; c = c > 1.f ? c : 1.f;
  eabf[(size_t)E * EDIM + i] = f32_to_bf16(lsum[i] / c);
}

// ------------------------------- WMMA GEMM ---------------------------------
// Y[M,128] = X[M,K](bf16) @ W[128,K](bf16)^T, f32 accumulate.
// Block = 256 threads = 8 waves = the 8 output-column tiles.
// Each block owns a 64-row supertile; each wave register-blocks 4 M-tiles.
// The 128x64 bf16 weight chunk (16 KB) is staged in LDS with
// global_load_async_to_lds_b128 (ASYNCcnt) and shared by all 8 waves.
// M-tile bases are CLAMPED (not branched) so the k-loop is branch-free:
// the last partial supertile recomputes the final tile redundantly, and
// only the stores are guarded.  Requires K % 64 == 0, M % 16 == 0.

union BF16Frag { v16bf v; unsigned u[8]; };

__global__ void gemm_bf16_wmma_lds(const unsigned short* __restrict__ X,
                                   const unsigned short* __restrict__ W,
                                   void* __restrict__ Y,
                                   int M, int K, int store_bf16) {
  __shared__ unsigned short shB[128 * KCHUNK];   // [col][64 K], 16 KB
  const int tid  = threadIdx.x;                  // 0..255
  const int wave = tid >> 5;                     // 0..7  == N-tile index
  const int lane = tid & 31;
  const int half = lane >> 4, l16 = lane & 15;
  const int m0   = blockIdx.x * 64;              // first row of supertile
  const int col  = wave * 16 + l16;              // output column of this lane
  const unsigned shBase = (unsigned)(uintptr_t)shB;  // LDS byte offset

  // clamped, branch-free tile bases (uniform across the k-loop)
  const unsigned short* Abase[4];
#pragma unroll
  for (int t = 0; t < 4; ++t) {
    int tb = m0 + t * 16;
    if (tb + 16 > M) tb = M - 16;                // always fully in-bounds
    Abase[t] = X + (size_t)(tb + l16) * K;
  }

  v8f acc[4] = {v8f{}, v8f{}, v8f{}, v8f{}};

  for (int k0 = 0; k0 < K; k0 += KCHUNK) {
    __syncthreads();   // previous chunk fully consumed before overwrite
    // ---- cooperative async load of W[:, k0:k0+64] into LDS (1024 x b128) --
#pragma unroll
    for (int r = 0; r < 4; ++r) {
      int i  = tid + 256 * r;              // uint4 index within chunk
      int cB = i >> 3;                     // weight row (output column)
      int kk = (i & 7) * 8;                // ushort offset within chunk row
      unsigned ldsOff = shBase + (unsigned)i * 16u;
      const unsigned short* g = W + (size_t)cB * K + k0 + kk;
      asm volatile("global_load_async_to_lds_b128 %0, %1, off"
                   :: "v"(ldsOff), "v"((unsigned long long)(uintptr_t)g)
                   : "memory");
    }
    __builtin_prefetch(Abase[0] + k0 + KCHUNK, 0, 1);   // next A chunk
    asm volatile("s_wait_asynccnt 0x0" ::: "memory");
    __syncthreads();

    // ---- two 16x16x32 WMMA k-steps against the LDS chunk (branch-free) ---
#pragma unroll
    for (int s = 0; s < 2; ++s) {
      BF16Frag b;
      {  // B fragment: 16 contiguous ushorts from LDS (2 x ds_load_b128)
        const uint4* bp = (const uint4*)&shB[col * KCHUNK + 32 * s + 16 * half];
        uint4* bu = (uint4*)b.u;
        bu[0] = bp[0];
        bu[1] = bp[1];
      }
#pragma unroll
      for (int t = 0; t < 4; ++t) {
        const unsigned short* Arow = Abase[t] + k0 + 32 * s;
        BF16Frag a;
#pragma unroll
        for (int j = 0; j < 8; ++j) {
          // A layout (contiguous, matches B): lanes0-15 K{0-15}, lanes16-31 K{16-31}
          int ka = 16 * half + 2 * j;
          a.u[j] = *(const unsigned*)(Arow + ka);
        }
        acc[t] = __builtin_amdgcn_wmma_f32_16x16x32_bf16(
            /*neg_a=*/false, a.v, /*neg_b=*/false, b.v,
            /*c_mod=*/(short)0, acc[t], /*reuse_a=*/false, /*reuse_b=*/false);
      }
    }
  }

  // ---- store D tiles (guard only here) ----
#pragma unroll
  for (int t = 0; t < 4; ++t) {
    int rbase = m0 + t * 16;
    if (rbase >= M) continue;
    if (store_bf16) {
      unsigned short* Yb = (unsigned short*)Y;
#pragma unroll
      for (int r = 0; r < 8; ++r)
        Yb[(size_t)(rbase + r + 8 * half) * HID + col] = f32_to_bf16(acc[t][r]);
    } else {
      float* Yf = (float*)Y;
#pragma unroll
      for (int r = 0; r < 8; ++r)
        Yf[(size_t)(rbase + r + 8 * half) * HID + col] = acc[t][r];
    }
  }
}

// ------------------------------ edge kernels -------------------------------
// score[e] = leaky_relu(xl[src] + xr[dst] + ee[e]) . att   (one wave / edge)
__global__ void edge_score_kernel(const float* __restrict__ xl, const float* __restrict__ xr,
                                  const unsigned short* __restrict__ ee,
                                  const int* __restrict__ srcf, const int* __restrict__ dstf,
                                  const float* __restrict__ att,
                                  float* __restrict__ score, int E2) {
  int e = blockIdx.x * (blockDim.x >> 5) + (threadIdx.x >> 5);
  if (e >= E2) return;
  int lane = threadIdx.x & 31;
  int s = srcf[e], d = dstf[e];
  const float4 xls = *(const float4*)(xl + (size_t)s * HID + lane * 4);
  const float4 xrs = *(const float4*)(xr + (size_t)d * HID + lane * 4);
  const unsigned* ep = (const unsigned*)(ee + (size_t)e * HID + lane * 4);
  unsigned e0 = ep[0], e1 = ep[1];
  const float4 at = *(const float4*)(att + lane * 4);
  float acc = lrelu(xls.x + xrs.x + bf16_to_f32(e0 & 0xffffu)) * at.x
            + lrelu(xls.y + xrs.y + bf16_to_f32(e0 >> 16))     * at.y
            + lrelu(xls.z + xrs.z + bf16_to_f32(e1 & 0xffffu)) * at.z
            + lrelu(xls.w + xrs.w + bf16_to_f32(e1 >> 16))     * at.w;
#pragma unroll
  for (int off = 16; off; off >>= 1) acc += __shfl_xor(acc, off, 32);
  if (lane == 0) score[e] = acc;
}

__global__ void seg_max_kernel(const float* __restrict__ score, const int* __restrict__ dstf,
                               unsigned* __restrict__ menc, int E2) {
  int e = blockIdx.x * blockDim.x + threadIdx.x;
  if (e < E2) atomicMax(&menc[dstf[e]], f2mono(score[e]));
}

__global__ void exp_den_kernel(const float* __restrict__ score, const int* __restrict__ dstf,
                               const unsigned* __restrict__ menc,
                               float* __restrict__ exs, float* __restrict__ den, int E2) {
  int e = blockIdx.x * blockDim.x + threadIdx.x;
  if (e >= E2) return;
  int d = dstf[e];
  float m = mono2f(menc[d]);
  if (!isfinite(m)) m = 0.f;
  float ex = __expf(score[e] - m);
  exs[e] = ex;
  atomicAdd(&den[d], ex);
}

// agg[dst] += alpha * xl[src]   (one wave / edge, 4 channels / lane)
__global__ void aggregate_kernel(const float* __restrict__ xl,
                                 const int* __restrict__ srcf, const int* __restrict__ dstf,
                                 const float* __restrict__ exs, const float* __restrict__ den,
                                 float* __restrict__ agg, int E2) {
  int e = blockIdx.x * (blockDim.x >> 5) + (threadIdx.x >> 5);
  if (e >= E2) return;
  int lane = threadIdx.x & 31;
  int s = srcf[e], d = dstf[e];
  float alpha = exs[e] / (den[d] + 1e-16f);
  const float4 xv = *(const float4*)(xl + (size_t)s * HID + lane * 4);
  float* o = agg + (size_t)d * HID + lane * 4;
  atomicAdd(o + 0, alpha * xv.x);
  atomicAdd(o + 1, alpha * xv.y);
  atomicAdd(o + 2, alpha * xv.z);
  atomicAdd(o + 3, alpha * xv.w);
}

// ----------------------------- batchnorm/prelu -----------------------------

__global__ void bn_stats_kernel(const float* __restrict__ h, const float* __restrict__ bias,
                                float* __restrict__ mean, float* __restrict__ rstd, int N) {
  int ch = blockIdx.x;
  float bv = bias[ch];
  float s = 0.f, s2 = 0.f;
  for (int n = threadIdx.x; n < N; n += blockDim.x) {
    float v = h[(size_t)n * HID + ch] + bv;
    s += v; s2 += v * v;
  }
  __shared__ float ss[256], ss2[256];
  ss[threadIdx.x] = s; ss2[threadIdx.x] = s2;
  __syncthreads();
  for (int o = blockDim.x >> 1; o; o >>= 1) {
    if (threadIdx.x < (unsigned)o) { ss[threadIdx.x] += ss[threadIdx.x + o];
                                     ss2[threadIdx.x] += ss2[threadIdx.x + o]; }
    __syncthreads();
  }
  if (threadIdx.x == 0) {
    float m = ss[0] / (float)N;
    float var = ss2[0] / (float)N - m * m;
    mean[ch] = m;
    rstd[ch] = rsqrtf(var + BN_EPS);
  }
}

__global__ void bn_apply_kernel(const float* __restrict__ agg, const float* __restrict__ bias,
                                const float* __restrict__ mean, const float* __restrict__ rstd,
                                const float* __restrict__ g, const float* __restrict__ b,
                                const float* __restrict__ pr,
                                float* __restrict__ hout, int N) {
  size_t i = (size_t)blockIdx.x * blockDim.x + threadIdx.x;
  if (i >= (size_t)N * HID) return;
  int ch = (int)(i & (HID - 1));
  float a = pr[0];
  float v = agg[i] + bias[ch];
  v = g[ch] * (v - mean[ch]) * rstd[ch] + b[ch];
  hout[i] = v >= 0.f ? v : a * v;
}

__global__ void bias_add_kernel(const float* __restrict__ agg, const float* __restrict__ bias,
                                float* __restrict__ hout, int N) {
  size_t i = (size_t)blockIdx.x * blockDim.x + threadIdx.x;
  if (i >= (size_t)N * HID) return;
  hout[i] = agg[i] + bias[(int)(i & (HID - 1))];
}

// ------------------------------ pooling + MLP ------------------------------

__global__ void pool_kernel(const float* __restrict__ h, const int* __restrict__ batch,
                            float* __restrict__ pooled, int N, int colOff) {
  int n = blockIdx.x;
  int c = threadIdx.x;
  if (n >= N) return;
  atomicAdd(&pooled[(size_t)batch[n] * (4 * HID) + colOff + c], h[(size_t)n * HID + c]);
}

// single block, 128 threads: bias + BN + PReLU + final linear + softmax + argmax
__global__ void mlp_finalize_kernel(const float* __restrict__ z0, const float* __restrict__ b1,
                                    const float* __restrict__ g, const float* __restrict__ bb,
                                    const float* __restrict__ pr,
                                    const float* __restrict__ W2, const float* __restrict__ b2,
                                    float* __restrict__ out) {
  __shared__ float z[NGRAPH * HID];
  __shared__ float logits[NGRAPH * 2];
  int tid = threadIdx.x;   // 0..127, tid == channel
  for (int r = 0; r < NGRAPH; ++r)
    z[r * HID + tid] = z0[r * HID + tid] + b1[tid];
  __syncthreads();
  float s = 0.f, s2 = 0.f;
  for (int r = 0; r < NGRAPH; ++r) { float v = z[r * HID + tid]; s += v; s2 += v * v; }
  float m = s / (float)NGRAPH;
  float var = s2 / (float)NGRAPH - m * m;
  float rs = rsqrtf(var + BN_EPS);
  float a = pr[0];
  for (int r = 0; r < NGRAPH; ++r) {
    float v = g[tid] * (z[r * HID + tid] - m) * rs + bb[tid];
    z[r * HID + tid] = v >= 0.f ? v : a * v;
  }
  __syncthreads();
  if (tid < NGRAPH * 2) {
    int r = tid >> 1, o = tid & 1;
    float acc = b2[o];
    for (int c = 0; c < HID; ++c) acc += z[r * HID + c] * W2[o * HID + c];
    logits[tid] = acc;
  }
  __syncthreads();
  if (tid < NGRAPH) {
    float l0 = logits[tid * 2], l1 = logits[tid * 2 + 1];
    float mx = fmaxf(l0, l1);
    float e0 = __expf(l0 - mx), e1 = __expf(l1 - mx);
    float inv = 1.f / (e0 + e1);
    out[tid * 2]     = l0;                  // logits
    out[tid * 2 + 1] = l1;
    out[64 + tid * 2]     = e0 * inv;       // prob
    out[64 + tid * 2 + 1] = e1 * inv;
    out[128 + tid] = (l1 > l0) ? 1.f : 0.f; // pred
  }
}

// -------------------------------- host side --------------------------------

extern "C" void kernel_launch(void* const* d_in, const int* in_sizes, int n_in,
                              void* d_out, int out_size, void* d_ws, size_t ws_size,
                              hipStream_t stream) {
  // ---- inputs (JAX pytree: dict keys in sorted order) ----
  const float* x          = (const float*)d_in[0];
  const int*   edge_index = (const int*)  d_in[1];
  const float* edge_attr  = (const float*)d_in[2];
  const int*   batch      = (const int*)  d_in[3];
  const float* bnb[3] = { (const float*)d_in[4], (const float*)d_in[6], (const float*)d_in[8] };
  const float* bng[3] = { (const float*)d_in[5], (const float*)d_in[7], (const float*)d_in[9] };
  const float *We[4], *Wl[4], *Wr[4], *att[4], *cbias[4];
  for (int l = 0; l < 4; ++l) {               // conv dict sorted: We, Wl, Wr, att, bias
    We[l]    = (const float*)d_in[10 + 5 * l + 0];
    Wl[l]    = (const float*)d_in[10 + 5 * l + 1];
    Wr[l]    = (const float*)d_in[10 + 5 * l + 2];
    att[l]   = (const float*)d_in[10 + 5 * l + 3];
    cbias[l] = (const float*)d_in[10 + 5 * l + 4];
  }
  const float* mlp_W1  = (const float*)d_in[30];
  const float* mlp_W2  = (const float*)d_in[31];
  const float* mlp_b1  = (const float*)d_in[32];
  const float* mlp_b2  = (const float*)d_in[33];
  const float* mlp_bnb = (const float*)d_in[34];
  const float* mlp_bng = (const float*)d_in[35];
  const float* mlp_pr  = (const float*)d_in[36];
  const float* pr[3]   = { (const float*)d_in[37], (const float*)d_in[38], (const float*)d_in[39] };

  const int E  = in_sizes[1] / 2;       // 400000
  const int N  = in_sizes[3];           // 50000
  const int E2 = E + N;                 // 450000 (with self loops)
  const int Kin[4] = { HID, HID, 2 * HID, 3 * HID };

  // ---- workspace bump allocator ----
  uintptr_t wp = (uintptr_t)d_ws;
  auto alloc = [&](size_t bytes) -> void* {
    uintptr_t r = (wp + 255) & ~(uintptr_t)255;
    wp = r + bytes;
    return (void*)r;
  };
  int*            srcf     = (int*)           alloc((size_t)E2 * 4);
  int*            dstf     = (int*)           alloc((size_t)E2 * 4);
  unsigned short* ea_bf    = (unsigned short*)alloc((size_t)E2 * EDIM * 2);
  float*          cnt      = (float*)         alloc((size_t)N * 4);
  float*          lsum     = (float*)         alloc((size_t)N * EDIM * 4);
  unsigned short* act_bf   = (unsigned short*)alloc((size_t)N * (3 * HID) * 2);
  float*          xl       = (float*)         alloc((size_t)N * HID * 4);
  float*          xr       = (float*)         alloc((size_t)N * HID * 4);
  unsigned short* ee_bf    = (unsigned short*)alloc((size_t)E2 * HID * 2);
  float*          score    = (float*)         alloc((size_t)E2 * 4);
  float*          exs      = (float*)         alloc((size_t)E2 * 4);
  unsigned*       menc     = (unsigned*)      alloc((size_t)N * 4);
  float*          den      = (float*)         alloc((size_t)N * 4);
  float*          agg      = (float*)         alloc((size_t)N * HID * 4);
  float*          mean     = (float*)         alloc(HID * 4);
  float*          rstd     = (float*)         alloc(HID * 4);
  float*          h[4];
  for (int l = 0; l < 4; ++l) h[l] = (float*)alloc((size_t)N * HID * 4);
  float*          pooled   = (float*)         alloc((size_t)NGRAPH * 4 * HID * 4);
  unsigned short* pooledbf = (unsigned short*)alloc((size_t)NGRAPH * 4 * HID * 2);
  float*          z0       = (float*)         alloc((size_t)NGRAPH * HID * 4);
  unsigned short* Wl_bf[4], *Wr_bf[4], *We_bf[4], *W1_bf;
  for (int l = 0; l < 4; ++l) {
    Wl_bf[l] = (unsigned short*)alloc((size_t)HID * Kin[l] * 2);
    Wr_bf[l] = (unsigned short*)alloc((size_t)HID * Kin[l] * 2);
    We_bf[l] = (unsigned short*)alloc((size_t)HID * EDIM * 2);
  }
  W1_bf = (unsigned short*)alloc((size_t)HID * 4 * HID * 2);
  (void)ws_size; (void)n_in; (void)out_size;

  auto blocks1 = [](size_t n) { return (unsigned)((n + 255) / 256); };
  auto fill = [&](void* p, unsigned v, size_t n) {
    fill_u32_kernel<<<blocks1(n), 256, 0, stream>>>((unsigned*)p, v, n);
  };
  auto cvt = [&](const float* s, unsigned short* d, size_t n) {
    f32_to_bf16_kernel<<<blocks1(n), 256, 0, stream>>>(s, d, n);
  };
  auto gemm = [&](const unsigned short* A, const unsigned short* B, void* Y,
                  int M, int K, int store_bf) {   // NO is always 128 (HID)
    gemm_bf16_wmma_lds<<<(M + 63) / 64, 256, 0, stream>>>(A, B, Y, M, K, store_bf);
  };

  // ---- preprocessing: self loops, edge-attr mean, bf16 weights ----
  build_edges_kernel<<<blocks1(E2), 256, 0, stream>>>(edge_index, srcf, dstf, E, N);
  cvt(edge_attr, ea_bf, (size_t)E * EDIM);
  fill(cnt, 0u, N);
  fill(lsum, 0u, (size_t)N * EDIM);
  cnt_kernel<<<blocks1(E), 256, 0, stream>>>(edge_index, cnt, E);
  loop_sum_kernel<<<(E + 7) / 8, 256, 0, stream>>>(edge_index, edge_attr, lsum, E);
  loopdiv_kernel<<<blocks1((size_t)N * EDIM), 256, 0, stream>>>(lsum, cnt, ea_bf, N, E);
  for (int l = 0; l < 4; ++l) {
    cvt(Wl[l], Wl_bf[l], (size_t)HID * Kin[l]);
    cvt(Wr[l], Wr_bf[l], (size_t)HID * Kin[l]);
    cvt(We[l], We_bf[l], (size_t)HID * EDIM);
  }
  cvt(mlp_W1, W1_bf, (size_t)HID * 4 * HID);

  // layer-1 input = x
  cat_bf16_kernel<<<blocks1((size_t)N * HID), 256, 0, stream>>>(x, act_bf, N, HID, HID, 0);

  // ---- four GATv2 layers ----
  for (int l = 0; l < 4; ++l) {
    int K = Kin[l];
    gemm(act_bf, Wl_bf[l], xl, N, K, 0);               // xl = act @ Wl^T  (WMMA)
    gemm(act_bf, Wr_bf[l], xr, N, K, 0);               // xr = act @ Wr^T  (WMMA)
    gemm(ea_bf, We_bf[l], ee_bf, E2, EDIM, 1);         // ee = ea  @ We^T  (WMMA, bf16 out)

    edge_score_kernel<<<(E2 + 7) / 8, 256, 0, stream>>>(xl, xr, ee_bf, srcf, dstf,
                                                        att[l], score, E2);
    fill(menc, 0x007FFFFFu, N);                        // f2mono(-inf)
    seg_max_kernel<<<blocks1(E2), 256, 0, stream>>>(score, dstf, menc, E2);
    fill(den, 0u, N);
    exp_den_kernel<<<blocks1(E2), 256, 0, stream>>>(score, dstf, menc, exs, den, E2);
    fill(agg, 0u, (size_t)N * HID);
    aggregate_kernel<<<(E2 + 7) / 8, 256, 0, stream>>>(xl, srcf, dstf, exs, den, agg, E2);

    if (l < 3) {
      bn_stats_kernel<<<HID, 256, 0, stream>>>(agg, cbias[l], mean, rstd, N);
      bn_apply_kernel<<<blocks1((size_t)N * HID), 256, 0, stream>>>(
          agg, cbias[l], mean, rstd, bng[l], bnb[l], pr[l], h[l], N);
    } else {
      bias_add_kernel<<<blocks1((size_t)N * HID), 256, 0, stream>>>(agg, cbias[3], h[3], N);
    }

    // build next layer's bf16 input (concat of earlier h's)
    if (l == 0) {
      cat_bf16_kernel<<<blocks1((size_t)N * HID), 256, 0, stream>>>(h[0], act_bf, N, HID, HID, 0);
    } else if (l == 1) {
      cat_bf16_kernel<<<blocks1((size_t)N * HID), 256, 0, stream>>>(h[0], act_bf, N, HID, 2 * HID, 0);
      cat_bf16_kernel<<<blocks1((size_t)N * HID), 256, 0, stream>>>(h[1], act_bf, N, HID, 2 * HID, HID);
    } else if (l == 2) {
      cat_bf16_kernel<<<blocks1((size_t)N * HID), 256, 0, stream>>>(h[0], act_bf, N, HID, 3 * HID, 0);
      cat_bf16_kernel<<<blocks1((size_t)N * HID), 256, 0, stream>>>(h[1], act_bf, N, HID, 3 * HID, HID);
      cat_bf16_kernel<<<blocks1((size_t)N * HID), 256, 0, stream>>>(h[2], act_bf, N, HID, 3 * HID, 2 * HID);
    }
  }

  // ---- global_add_pool over 4H concat ----
  fill(pooled, 0u, (size_t)NGRAPH * 4 * HID);
  for (int l = 0; l < 4; ++l)
    pool_kernel<<<N, HID, 0, stream>>>(h[l], batch, pooled, N, l * HID);

  // ---- MLP head: z0 = pooled @ W1^T (WMMA), then fused finalize ----
  cvt(pooled, pooledbf, (size_t)NGRAPH * 4 * HID);
  gemm(pooledbf, W1_bf, z0, NGRAPH, 4 * HID, 0);
  mlp_finalize_kernel<<<1, HID, 0, stream>>>(z0, mlp_b1, mlp_bng, mlp_bnb, mlp_pr,
                                             mlp_W2, mlp_b2, (float*)d_out);
}
